// NN_66305705115875
// MI455X (gfx1250) — matrix operand order repaired and checked by reading, not compiled, not run on hardware
//
#include <hip/hip_runtime.h>

// ---------------------------------------------------------------------------
// Types for CDNA5 WMMA (wave32): v_wmma_f32_16x16x32_f16
// ---------------------------------------------------------------------------
typedef __attribute__((ext_vector_type(16))) _Float16      v16h;
typedef __attribute__((ext_vector_type(8)))  float         v8f;
typedef __attribute__((ext_vector_type(4)))  unsigned int  u32x4;
typedef __attribute__((ext_vector_type(2)))  unsigned int  u32x2;
typedef __attribute__((ext_vector_type(4)))  float         f32x4;
typedef int i32x4v __attribute__((vector_size(16)));   // matches async builtin param

union Frag {
    unsigned u[8];
    u32x4    q[2];
    v16h     v;
};

#define GLOBAL_AS __attribute__((address_space(1)))
#define LDS_AS    __attribute__((address_space(3)))

// CDNA5 async global->LDS copies (ASYNCcnt) if this toolchain exposes them.
#if defined(__AMDGCN__) &&                                                \
    __has_builtin(__builtin_amdgcn_global_load_async_to_lds_b128) &&      \
    __has_builtin(__builtin_amdgcn_global_load_async_to_lds_b32) &&       \
    __has_builtin(__builtin_amdgcn_s_wait_asynccnt)
#define USE_ASYNC 1
#else
#define USE_ASYNC 0
#endif

// f32 pair -> packed f16 (single v_cvt_pk_rtz_f16_f32 on device)
#if defined(__AMDGCN__) && __has_builtin(__builtin_amdgcn_cvt_pkrtz)
__device__ __forceinline__ unsigned pk2(float a, float b) {
    return __builtin_bit_cast(unsigned, __builtin_amdgcn_cvt_pkrtz(a, b));
}
#else
__device__ __forceinline__ unsigned pk2(float a, float b) {
    union { _Float16 h[2]; unsigned u; } cv;
    cv.h[0] = (_Float16)a; cv.h[1] = (_Float16)b;
    return cv.u;
}
#endif

__device__ __forceinline__ unsigned short f2h(float f) {
    union { _Float16 h; unsigned short s; } cv;
    cv.h = (_Float16)f;
    return cv.s;
}

// order-preserving float <-> uint mapping for atomic min/max
__device__ __forceinline__ unsigned f2ord(float f) {
    unsigned u = __float_as_uint(f);
    return (u & 0x80000000u) ? ~u : (u | 0x80000000u);
}
__device__ __forceinline__ float ord2f(unsigned o) {
    unsigned u = (o & 0x80000000u) ? (o & 0x7FFFFFFFu) : ~o;
    return __uint_as_float(u);
}

#if USE_ASYNC
__device__ __forceinline__ void acp128(const float* g, float* l) {
    __builtin_amdgcn_global_load_async_to_lds_b128(
        (GLOBAL_AS i32x4v*)(GLOBAL_AS void*)g,
        (LDS_AS i32x4v*)(LDS_AS void*)l, 0, 0);
}
__device__ __forceinline__ void acp32(const float* g, float* l) {
    __builtin_amdgcn_global_load_async_to_lds_b32(
        (GLOBAL_AS int*)(GLOBAL_AS void*)g,
        (LDS_AS int*)(LDS_AS void*)l, 0, 0);
}
#endif

// Pack 16 consecutive f32 (LDS) -> one f16 WMMA B-fragment
__device__ __forceinline__ void pack16(Frag& f, const float* p) {
    f32x4 x0 = *(const f32x4*)(p);
    f32x4 x1 = *(const f32x4*)(p + 4);
    f32x4 x2 = *(const f32x4*)(p + 8);
    f32x4 x3 = *(const f32x4*)(p + 12);
    f.u[0] = pk2(x0[0], x0[1]); f.u[1] = pk2(x0[2], x0[3]);
    f.u[2] = pk2(x1[0], x1[1]); f.u[3] = pk2(x1[2], x1[3]);
    f.u[4] = pk2(x2[0], x2[1]); f.u[5] = pk2(x2[2], x2[3]);
    f.u[6] = pk2(x3[0], x3[1]); f.u[7] = pk2(x3[2], x3[3]);
}
__device__ __forceinline__ void packA(Frag& f, const float* row, int half) {
    // A frag ks: {16h..16h+15}
    f32x4 x0 = *(const f32x4*)(row + 16 * half);
    f32x4 x1 = *(const f32x4*)(row + 16 * half + 4);
    f32x4 x2 = *(const f32x4*)(row + 16 * half + 8);
    f32x4 x3 = *(const f32x4*)(row + 16 * half + 12);
    f.u[0] = pk2(x0[0], x0[1]); f.u[1] = pk2(x0[2], x0[3]);
    f.u[2] = pk2(x1[0], x1[1]); f.u[3] = pk2(x1[2], x1[3]);
    f.u[4] = pk2(x2[0], x2[1]); f.u[5] = pk2(x2[2], x2[3]);
    f.u[6] = pk2(x3[0], x3[1]); f.u[7] = pk2(x3[2], x3[3]);
}

// ---------------------------------------------------------------------------
// Forward GEMM: Y[M,N] = relu(A[M,K] @ Wt[N,K]^T + bias), f16 WMMA, f32 acc.
// Block: 256 thr (8 waves), tile 128(M) x 64(N), K step 32.
// Async path: double-buffered global->LDS async b128 copies (f32 tiles,
// pitch 36), f16 packing at fragment time. Fallback: sync f16 staging.
// ---------------------------------------------------------------------------
__global__ __launch_bounds__(256) void k_gemm_fwd(
    const float* __restrict__ A, const float* __restrict__ Wt,
    const float* __restrict__ bias, float* __restrict__ Y, int N, int K)
{
    const int tid   = threadIdx.x;
    const int lane  = tid & 31;
    const int wave  = tid >> 5;
    const int waveM = wave & 3;
    const int waveN = wave >> 2;
    const int half  = lane >> 4;
    const int mr    = lane & 15;
    const int m0 = blockIdx.x * 128;
    const int n0 = blockIdx.y * 64;

    v8f acc[2][2] = {};

#if USE_ASYNC
    __shared__ float sA[2][128 * 36];
    __shared__ float sB[2][64 * 36];
    const int arow = tid >> 1;          // 0..127
    const int akb  = (tid & 1) * 16;    // 0 or 16
    const int brow = tid >> 2;          // 0..63
    const int bkb  = (tid & 3) * 8;     // 0..24
    const int nCh  = K >> 5;

    // prologue: issue chunk 0 into buffer 0
    {
        const float* ga = A + (size_t)(m0 + arow) * K + akb;
        const float* gb = Wt + (size_t)(n0 + brow) * K + bkb;
#pragma unroll
        for (int i = 0; i < 4; ++i) acp128(ga + 4 * i, &sA[0][arow * 36 + akb + 4 * i]);
#pragma unroll
        for (int i = 0; i < 2; ++i) acp128(gb + 4 * i, &sB[0][brow * 36 + bkb + 4 * i]);
    }
    for (int c = 0; c < nCh; ++c) {
        const int cur = c & 1;
        if (c + 1 < nCh) {
            const int k1 = (c + 1) << 5;
            const float* ga = A + (size_t)(m0 + arow) * K + k1 + akb;
            const float* gb = Wt + (size_t)(n0 + brow) * K + k1 + bkb;
#pragma unroll
            for (int i = 0; i < 4; ++i) acp128(ga + 4 * i, &sA[cur ^ 1][arow * 36 + akb + 4 * i]);
#pragma unroll
            for (int i = 0; i < 2; ++i) acp128(gb + 4 * i, &sB[cur ^ 1][brow * 36 + bkb + 4 * i]);
            __builtin_amdgcn_s_wait_asynccnt(6);   // chunk c done; c+1 in flight
        } else {
            __builtin_amdgcn_s_wait_asynccnt(0);   // final chunk fully landed
        }
        __syncthreads();

        Frag aF[2], bF[2];
#pragma unroll
        for (int i = 0; i < 2; ++i)
            packA(aF[i], &sA[cur][(waveM * 32 + 16 * i + mr) * 36], half);
#pragma unroll
        for (int j = 0; j < 2; ++j)
            pack16(bF[j], &sB[cur][(waveN * 32 + 16 * j + mr) * 36 + 16 * half]);
#pragma unroll
        for (int i = 0; i < 2; ++i)
#pragma unroll
            for (int j = 0; j < 2; ++j)
                acc[i][j] = __builtin_amdgcn_wmma_f32_16x16x32_f16(
                    false, aF[i].v, false, bF[j].v, (short)0, acc[i][j], false, false);
        __syncthreads();
    }
#else
    __shared__ unsigned short sA[128 * 40];
    __shared__ unsigned short sB[64 * 40];
    const int arow = tid >> 1;
    const int akb  = (tid & 1) * 16;
    const int brow = tid >> 2;
    const int bkb  = (tid & 3) * 8;

    for (int k0 = 0; k0 < K; k0 += 32) {
        __syncthreads();
        {
            const float* ga = A + (size_t)(m0 + arow) * K + k0 + akb;
#pragma unroll
            for (int i = 0; i < 4; ++i) {
                f32x4 v = *(const f32x4*)(ga + 4 * i);
                u32x2 p; p.x = pk2(v[0], v[1]); p.y = pk2(v[2], v[3]);
                *(u32x2*)(sA + arow * 40 + akb + 4 * i) = p;
            }
            if (k0 + 32 < K) __builtin_prefetch(ga + 32, 0, 1);
        }
        {
            const float* gb = Wt + (size_t)(n0 + brow) * K + k0 + bkb;
#pragma unroll
            for (int i = 0; i < 2; ++i) {
                f32x4 v = *(const f32x4*)(gb + 4 * i);
                u32x2 p; p.x = pk2(v[0], v[1]); p.y = pk2(v[2], v[3]);
                *(u32x2*)(sB + brow * 40 + bkb + 4 * i) = p;
            }
        }
        __syncthreads();

        Frag aF[2], bF[2];
#pragma unroll
        for (int i = 0; i < 2; ++i) {
            const unsigned short* pa = sA + (waveM * 32 + 16 * i + mr) * 40;
            aF[i].q[0] = *(const u32x4*)(pa + 16 * half);
            aF[i].q[1] = *(const u32x4*)(pa + 16 * half + 8);
        }
#pragma unroll
        for (int j = 0; j < 2; ++j) {
            const unsigned short* pb = sB + (waveN * 32 + 16 * j + mr) * 40 + 16 * half;
            bF[j].q[0] = *(const u32x4*)(pb);
            bF[j].q[1] = *(const u32x4*)(pb + 8);
        }
#pragma unroll
        for (int i = 0; i < 2; ++i)
#pragma unroll
            for (int j = 0; j < 2; ++j)
                acc[i][j] = __builtin_amdgcn_wmma_f32_16x16x32_f16(
                    false, aF[i].v, false, bF[j].v, (short)0, acc[i][j], false, false);
    }
#endif

    // epilogue: bias + relu; C layout lane: col=mr, VGPR v -> row v+8*half
#pragma unroll
    for (int j = 0; j < 2; ++j) {
        const int col = n0 + waveN * 32 + 16 * j + mr;
        const float bv = bias[col];
#pragma unroll
        for (int i = 0; i < 2; ++i)
#pragma unroll
            for (int v = 0; v < 8; ++v) {
                const int row = m0 + waveM * 32 + 16 * i + v + 8 * half;
                float y = acc[i][j][v] + bv;
                Y[(size_t)row * N + col] = fmaxf(y, 0.0f);
            }
    }
}

// ---------------------------------------------------------------------------
// Split-K correlation GEMM: C[M,N] += P[K,M]^T @ Q[K,N]  (fp32 atomic acc).
// Block: 256 thr, tile 64x64, K step 32, K sliced along gridDim.z.
// Async path: per-lane-addressed async b32 copies scatter-transpose K-major
// tiles into [m][k]/[n][k] f32 LDS tiles, double buffered.
// ---------------------------------------------------------------------------
__global__ __launch_bounds__(256) void k_gemm_tn(
    const float* __restrict__ P, const float* __restrict__ Q,
    float* __restrict__ C, int M, int N, int kPerSlice)
{
    const int tid   = threadIdx.x;
    const int lane  = tid & 31;
    const int wave  = tid >> 5;
    const int waveM = wave & 1;
    const int waveN = wave >> 1;
    const int half  = lane >> 4;
    const int mr    = lane & 15;
    const int m0 = blockIdx.x * 64;
    const int n0 = blockIdx.y * 64;
    const int kStart = blockIdx.z * kPerSlice;

    const int skk = tid >> 3;        // 0..31 (k within chunk)
    const int smb = (tid & 7) * 8;   // 0..56 (m/n base)

    v8f acc[2] = {};

#if USE_ASYNC
    __shared__ float sP[2][64 * 36];
    __shared__ float sQ[2][64 * 36];
    const int nCh = kPerSlice >> 5;

    {
        const float* gp = P + (size_t)(kStart + skk) * M + m0 + smb;
        const float* gq = Q + (size_t)(kStart + skk) * N + n0 + smb;
#pragma unroll
        for (int t = 0; t < 8; ++t) acp32(gp + t, &sP[0][(smb + t) * 36 + skk]);
#pragma unroll
        for (int t = 0; t < 8; ++t) acp32(gq + t, &sQ[0][(smb + t) * 36 + skk]);
    }
    for (int c = 0; c < nCh; ++c) {
        const int cur = c & 1;
        if (c + 1 < nCh) {
            const int k1 = kStart + ((c + 1) << 5);
            const float* gp = P + (size_t)(k1 + skk) * M + m0 + smb;
            const float* gq = Q + (size_t)(k1 + skk) * N + n0 + smb;
#pragma unroll
            for (int t = 0; t < 8; ++t) acp32(gp + t, &sP[cur ^ 1][(smb + t) * 36 + skk]);
#pragma unroll
            for (int t = 0; t < 8; ++t) acp32(gq + t, &sQ[cur ^ 1][(smb + t) * 36 + skk]);
            __builtin_amdgcn_s_wait_asynccnt(16);
        } else {
            __builtin_amdgcn_s_wait_asynccnt(0);
        }
        __syncthreads();

        Frag aF[2], bF;
#pragma unroll
        for (int i = 0; i < 2; ++i)
            packA(aF[i], &sP[cur][(waveM * 32 + 16 * i + mr) * 36], half);
        pack16(bF, &sQ[cur][(waveN * 16 + mr) * 36 + 16 * half]);
#pragma unroll
        for (int i = 0; i < 2; ++i)
            acc[i] = __builtin_amdgcn_wmma_f32_16x16x32_f16(
                false, aF[i].v, false, bF.v, (short)0, acc[i], false, false);
        __syncthreads();
    }
#else
    __shared__ unsigned short sP[64 * 40];
    __shared__ unsigned short sQ[64 * 40];

    for (int kc = 0; kc < kPerSlice; kc += 32) {
        const int k0 = kStart + kc;
        __syncthreads();
        {
            const float* gp = P + (size_t)(k0 + skk) * M + m0 + smb;
            f32x4 p0 = *(const f32x4*)(gp);
            f32x4 p1 = *(const f32x4*)(gp + 4);
#pragma unroll
            for (int t = 0; t < 4; ++t) sP[(smb + t) * 40 + skk] = f2h(p0[t]);
#pragma unroll
            for (int t = 0; t < 4; ++t) sP[(smb + 4 + t) * 40 + skk] = f2h(p1[t]);
            const float* gq = Q + (size_t)(k0 + skk) * N + n0 + smb;
            f32x4 q0 = *(const f32x4*)(gq);
            f32x4 q1 = *(const f32x4*)(gq + 4);
#pragma unroll
            for (int t = 0; t < 4; ++t) sQ[(smb + t) * 40 + skk] = f2h(q0[t]);
#pragma unroll
            for (int t = 0; t < 4; ++t) sQ[(smb + 4 + t) * 40 + skk] = f2h(q1[t]);
            if (kc + 32 < kPerSlice) {
                __builtin_prefetch(gp + (size_t)32 * M, 0, 1);
                __builtin_prefetch(gq + (size_t)32 * N, 0, 1);
            }
        }
        __syncthreads();

        Frag aF[2], bF;
#pragma unroll
        for (int i = 0; i < 2; ++i) {
            const unsigned short* pa = sP + (waveM * 32 + 16 * i + mr) * 40;
            aF[i].q[0] = *(const u32x4*)(pa + 16 * half);
            aF[i].q[1] = *(const u32x4*)(pa + 16 * half + 8);
        }
        {
            const unsigned short* pb = sQ + (waveN * 16 + mr) * 40 + 16 * half;
            bF.q[0] = *(const u32x4*)(pb);
            bF.q[1] = *(const u32x4*)(pb + 8);
        }
#pragma unroll
        for (int i = 0; i < 2; ++i)
            acc[i] = __builtin_amdgcn_wmma_f32_16x16x32_f16(
                false, aF[i].v, false, bF.v, (short)0, acc[i], false, false);
    }
#endif

    const int col = n0 + waveN * 16 + mr;
#pragma unroll
    for (int i = 0; i < 2; ++i)
#pragma unroll
        for (int v = 0; v < 8; ++v) {
            const int row = m0 + waveM * 32 + 16 * i + v + 8 * half;
            atomicAdd(&C[(size_t)row * N + col], acc[i][v]);
        }
}

// ---------------------------------------------------------------------------
// Column sums of Y[B,N] (N = 64/128/256, power of two) -> cs[N]
// ---------------------------------------------------------------------------
__global__ __launch_bounds__(256) void k_colsum(
    const float* __restrict__ Y, float* __restrict__ cs,
    int N, int nshift, int rowsPer)
{
    const int r0   = blockIdx.x * rowsPer;
    const int col  = threadIdx.x & (N - 1);
    const int rsub = threadIdx.x >> nshift;
    const int rstep = 256 >> nshift;
    float a = 0.0f;
    for (int r = r0 + rsub; r < r0 + rowsPer; r += rstep)
        a += Y[(size_t)r * N + col];
    atomicAdd(&cs[col], a);
}

// ---------------------------------------------------------------------------
// Winner mask from batch-averaged activity; losers = ascending non-winners.
// ---------------------------------------------------------------------------
__global__ void k_winner(const float* __restrict__ colsum, float Binv, int n, int kk,
                         float* __restrict__ winner, float* __restrict__ mask_out,
                         float* __restrict__ losers_out)
{
    __shared__ float avg[256];
    __shared__ float swin[256];
    const int c = threadIdx.x;
    avg[c] = colsum[c] * Binv;
    __syncthreads();
    const float v = avg[c];
    int rank = 0;
    for (int j = 0; j < n; ++j) {
        const float o = avg[j];
        rank += (o > v) || (o == v && j < c);   // stable top-k tie-break
    }
    const float w = (rank < kk) ? 1.0f : 0.0f;
    swin[c] = w;
    winner[c]   = w;
    mask_out[c] = w;
    __syncthreads();
    if (w == 0.0f) {
        int pos = 0;
        for (int j = 0; j < c; ++j) pos += (swin[j] == 0.0f);
        losers_out[pos] = (float)c;
    }
}

// ---------------------------------------------------------------------------
// Per-sample top-k membership: rowsel[r]=1 if r in top-k of ANY sample row.
// ---------------------------------------------------------------------------
__global__ __launch_bounds__(256) void k_rowsel(
    const float* __restrict__ Y, float* __restrict__ rowsel,
    int Bn, int N, int kk, int spb)
{
    __shared__ float row[256];
    const int base = blockIdx.x * spb;
    for (int s = 0; s < spb; ++s) {
        const int b = base + s;
        __syncthreads();
        if (b < Bn && (int)threadIdx.x < N)
            row[threadIdx.x] = Y[(size_t)b * N + threadIdx.x];
        __syncthreads();
        if (b < Bn && (int)threadIdx.x < N) {
            const float v = row[threadIdx.x];
            int rank = 0;
            for (int j = 0; j < N; ++j) {
                const float o = row[j];
                rank += (o > v) || (o == v && j < (int)threadIdx.x);
            }
            if (rank < kk) rowsel[threadIdx.x] = 1.0f;   // benign race: all write 1
        }
    }
}

// ---------------------------------------------------------------------------
// scale_raw[r,c] = (win[r]*G[r,c]/B - sum_{j<=r} win[r]win[j]YY[r,j]/B * W[j,c])
//                  * rowsel[r];  also global min/max via ordered-uint atomics.
// ---------------------------------------------------------------------------
__global__ __launch_bounds__(256) void k_scale_raw(
    const float* __restrict__ G, const float* __restrict__ YY,
    const float* __restrict__ W, const float* __restrict__ winner,
    const float* __restrict__ rowsel, float* __restrict__ scale,
    unsigned* __restrict__ mm, int n, int in, float Binv)
{
    __shared__ float lat[256];
    __shared__ float smin[256], smax[256];
    const int r = blockIdx.x;
    const int c = blockIdx.y * 256 + threadIdx.x;
    if ((int)threadIdx.x < n)
        lat[threadIdx.x] = winner[r] * winner[threadIdx.x] * Binv * YY[r * n + threadIdx.x];
    __syncthreads();
    float v = 0.0f;
    const bool act = (c < in);
    if (act) {
        float acc = 0.0f;
        for (int j = 0; j <= r; ++j) acc = fmaf(lat[j], W[(size_t)j * in + c], acc);
        v = (winner[r] * G[(size_t)r * in + c] * Binv - acc) * rowsel[r];
        scale[(size_t)r * in + c] = v;
    }
    smin[threadIdx.x] = act ? v :  3.4e38f;
    smax[threadIdx.x] = act ? v : -3.4e38f;
    __syncthreads();
    for (int s = 128; s > 0; s >>= 1) {
        if ((int)threadIdx.x < s) {
            smin[threadIdx.x] = fminf(smin[threadIdx.x], smin[threadIdx.x + s]);
            smax[threadIdx.x] = fmaxf(smax[threadIdx.x], smax[threadIdx.x + s]);
        }
        __syncthreads();
    }
    if (threadIdx.x == 0) {
        atomicMin(&mm[0], f2ord(smin[0]));
        atomicMax(&mm[1], f2ord(smax[0]));
    }
}

__global__ void k_normalize(float* __restrict__ s, int count, const unsigned* __restrict__ mm)
{
    const float mn  = ord2f(mm[0]);
    const float mx  = ord2f(mm[1]);
    const float inv = 1.0f / (mx - mn + 1e-8f);
    const int i = blockIdx.x * blockDim.x + threadIdx.x;
    if (i < count) s[i] = (s[i] - mn) * inv;
}

// ---------------------------------------------------------------------------
// Head: z = Y2 @ W3^T + b3 (10x64), softmax over 10, one sample per thread.
// ---------------------------------------------------------------------------
__global__ __launch_bounds__(256) void k_out3(
    const float* __restrict__ Y2, const float* __restrict__ W3,
    const float* __restrict__ b3, float* __restrict__ out)
{
    __shared__ float w[10 * 64];
    __shared__ float bb[10];
    for (int i = threadIdx.x; i < 640; i += 256) w[i] = W3[i];
    if (threadIdx.x < 10) bb[threadIdx.x] = b3[threadIdx.x];
    __syncthreads();
    const int b = blockIdx.x * 256 + threadIdx.x;
    const float* y = Y2 + (size_t)b * 64;
    float z[10];
    float zmax = -3.4e38f;
#pragma unroll
    for (int j = 0; j < 10; ++j) {
        float a = bb[j];
        for (int c = 0; c < 64; ++c) a = fmaf(y[c], w[j * 64 + c], a);
        z[j] = a;
        zmax = fmaxf(zmax, a);
    }
    float sum = 0.0f;
#pragma unroll
    for (int j = 0; j < 10; ++j) { z[j] = __expf(z[j] - zmax); sum += z[j]; }
    const float inv = 1.0f / sum;
#pragma unroll
    for (int j = 0; j < 10; ++j) out[(size_t)b * 10 + j] = z[j] * inv;
}

__global__ void k_zero(float* __restrict__ p, int n)
{
    const int i = blockIdx.x * blockDim.x + threadIdx.x;
    if (i < n) p[i] = 0.0f;
}

__global__ void k_initmm(unsigned* __restrict__ mm)
{
    const int i = threadIdx.x;
    if (i < 6) mm[i] = (i & 1) ? 0u : 0xFFFFFFFFu;
}

// ---------------------------------------------------------------------------
// Launcher
// ---------------------------------------------------------------------------
extern "C" void kernel_launch(void* const* d_in, const int* in_sizes, int n_in,
                              void* d_out, int out_size, void* d_ws, size_t ws_size,
                              hipStream_t stream)
{
    (void)in_sizes; (void)n_in; (void)out_size; (void)ws_size;
    const float* x  = (const float*)d_in[0];
    const float* W0 = (const float*)d_in[1];
    const float* b0 = (const float*)d_in[2];
    const float* W1 = (const float*)d_in[3];
    const float* b1 = (const float*)d_in[4];
    const float* W2 = (const float*)d_in[5];
    const float* b2 = (const float*)d_in[6];
    const float* W3 = (const float*)d_in[7];
    const float* b3 = (const float*)d_in[8];
    float* out = (float*)d_out;

    const int   Bn   = 65536;
    const float Binv = 1.0f / 65536.0f;

    // d_out float offsets (flattened tuple order)
    const size_t O_OUT = 0;
    const size_t O_S0 = 655360, O_S1 = 917504, O_S2 = 950272;
    const size_t O_L0 = 958464, O_L1 = 958592, O_L2 = 958656;
    const size_t O_M0 = 958688, O_M1 = 958944, O_M2 = 959072;

    // workspace carve-up (256B aligned)
    size_t off = 0;
    auto carve = [&](size_t bytes) {
        void* p = (char*)d_ws + off;
        off += (bytes + 255) & ~(size_t)255;
        return p;
    };
    float* Y0 = (float*)carve((size_t)Bn * 256 * 4);
    float* Y1 = (float*)carve((size_t)Bn * 128 * 4);
    float* Y2 = (float*)carve((size_t)Bn * 64 * 4);
    char* zbase = (char*)d_ws + off;
    float* G0  = (float*)carve(256 * 1024 * 4);
    float* G1  = (float*)carve(128 * 256 * 4);
    float* G2  = (float*)carve(64 * 128 * 4);
    float* YY0 = (float*)carve(256 * 256 * 4);
    float* YY1 = (float*)carve(128 * 128 * 4);
    float* YY2 = (float*)carve(64 * 64 * 4);
    float* cs0 = (float*)carve(256 * 4);
    float* cs1 = (float*)carve(128 * 4);
    float* cs2 = (float*)carve(64 * 4);
    float* rs0 = (float*)carve(256 * 4);
    float* rs1 = (float*)carve(128 * 4);
    float* rs2 = (float*)carve(64 * 4);
    const size_t zbytes = (size_t)((char*)d_ws + off - zbase);
    float* win0 = (float*)carve(256 * 4);
    float* win1 = (float*)carve(128 * 4);
    float* win2 = (float*)carve(64 * 4);
    unsigned* mm = (unsigned*)carve(64);

    const int zcount = (int)(zbytes / 4);
    k_zero<<<(zcount + 255) / 256, 256, 0, stream>>>((float*)zbase, zcount);
    k_initmm<<<1, 32, 0, stream>>>(mm);

    // ---- layer 0 : in=1024, n=256, k=128 ----
    k_gemm_fwd<<<dim3(512, 4), 256, 0, stream>>>(x, W0, b0, Y0, 256, 1024);
    k_colsum<<<256, 256, 0, stream>>>(Y0, cs0, 256, 8, 256);
    k_winner<<<1, 256, 0, stream>>>(cs0, Binv, 256, 128, win0, out + O_M0, out + O_L0);
    k_rowsel<<<2048, 256, 0, stream>>>(Y0, rs0, Bn, 256, 128, 32);
    k_gemm_tn<<<dim3(4, 16, 32), 256, 0, stream>>>(Y0, x,  G0,  256, 1024, 2048);
    k_gemm_tn<<<dim3(4, 4, 32),  256, 0, stream>>>(Y0, Y0, YY0, 256, 256,  2048);
    k_scale_raw<<<dim3(256, 4), 256, 0, stream>>>(G0, YY0, W0, win0, rs0,
                                                  out + O_S0, mm + 0, 256, 1024, Binv);

    // ---- layer 1 : in=256, n=128, k=64 ----
    k_gemm_fwd<<<dim3(512, 2), 256, 0, stream>>>(Y0, W1, b1, Y1, 128, 256);
    k_colsum<<<256, 256, 0, stream>>>(Y1, cs1, 128, 7, 256);
    k_winner<<<1, 128, 0, stream>>>(cs1, Binv, 128, 64, win1, out + O_M1, out + O_L1);
    k_rowsel<<<2048, 256, 0, stream>>>(Y1, rs1, Bn, 128, 64, 32);
    k_gemm_tn<<<dim3(2, 4, 32), 256, 0, stream>>>(Y1, Y0, G1,  128, 256, 2048);
    k_gemm_tn<<<dim3(2, 2, 32), 256, 0, stream>>>(Y1, Y1, YY1, 128, 128, 2048);
    k_scale_raw<<<dim3(128, 1), 256, 0, stream>>>(G1, YY1, W1, win1, rs1,
                                                  out + O_S1, mm + 2, 128, 256, Binv);

    // ---- layer 2 : in=128, n=64, k=32 ----
    k_gemm_fwd<<<dim3(512, 1), 256, 0, stream>>>(Y1, W2, b2, Y2, 64, 128);
    k_colsum<<<256, 256, 0, stream>>>(Y2, cs2, 64, 6, 256);
    k_winner<<<1, 64, 0, stream>>>(cs2, Binv, 64, 32, win2, out + O_M2, out + O_L2);
    k_rowsel<<<2048, 256, 0, stream>>>(Y2, rs2, Bn, 64, 32, 32);
    k_gemm_tn<<<dim3(1, 2, 32), 256, 0, stream>>>(Y2, Y1, G2,  64, 128, 2048);
    k_gemm_tn<<<dim3(1, 1, 32), 256, 0, stream>>>(Y2, Y2, YY2, 64, 64,  2048);
    k_scale_raw<<<dim3(64, 1), 256, 0, stream>>>(G2, YY2, W2, win2, rs2,
                                                 out + O_S2, mm + 4, 64, 128, Binv);

    // ---- min-max normalize scales ----
    k_normalize<<<(262144 + 255) / 256, 256, 0, stream>>>(out + O_S0, 262144, mm + 0);
    k_normalize<<<(32768 + 255) / 256, 256, 0, stream>>>(out + O_S1, 32768, mm + 2);
    k_normalize<<<(8192 + 255) / 256, 256, 0, stream>>>(out + O_S2, 8192, mm + 4);

    // ---- head + softmax ----
    k_out3<<<256, 256, 0, stream>>>(Y2, W3, b3, out + O_OUT);
}
